// CrossAttention_70677981823255
// MI455X (gfx1250) — matrix operand order repaired and checked by reading, not compiled, not run on hardware
//
#include <hip/hip_runtime.h>
#include <hip/hip_bf16.h>
#include <math.h>

typedef __attribute__((ext_vector_type(16))) __bf16 v16bf;
typedef __attribute__((ext_vector_type(8)))  __bf16 v8bf;
typedef __attribute__((ext_vector_type(8)))  float  v8f;

#define HEADS    8
#define DIM_HEAD 64
#define INNER    512
#define BATCH    8
#define CCH      320
#define NQ       4096
#define SKV      77
#define SPAD     96     // 3 x 32 (K-dim granularity for attn@V)
#define DKV      1024

#define SKST 72         // LDS row stride for K [s][d]   (144B = 36 banks, conflict-free)
#define SVST 104        // LDS row stride for V^T [d][s] (208B = 52 banks, conflict-free)
#define PST  104        // LDS row stride for P staging
#define AST  328        // LDS row stride for q_proj A staging (656B = 36 banks mod 64)

__device__ __forceinline__ v8f wmma_bf16(v16bf a, v16bf b, v8f c) {
  return __builtin_amdgcn_wmma_f32_16x16x32_bf16(false, a, false, b, (short)0, c, false, false);
}

__device__ __forceinline__ v16bf cat(v8bf lo, v8bf hi) {
  return __builtin_shufflevector(lo, hi, 0,1,2,3,4,5,6,7,8,9,10,11,12,13,14,15);
}

__device__ __forceinline__ v8bf cvt8(float4 a, float4 b) {
  v8bf r;
  r[0] = (__bf16)a.x; r[1] = (__bf16)a.y; r[2] = (__bf16)a.z; r[3] = (__bf16)a.w;
  r[4] = (__bf16)b.x; r[5] = (__bf16)b.y; r[6] = (__bf16)b.z; r[7] = (__bf16)b.w;
  return r;
}

// A-fragment (16-bit, 16x32): lane holds row M=lane16; element runs are
// K = {base+half*8 .. +7} and {base+16+half*8 .. +7}  -> two contiguous 16B loads.
// B-fragment (16-bit, 32x16): lane holds col N=lane16; elements are
// K = {base+half*8 .. +7} and {base+16+half*8 .. +7}  -> two contiguous 16B loads.
__device__ __forceinline__ v16bf load_a_bf16(const __bf16* row, int kbase, int half) {
  v8bf a0 = *(const v8bf*)(row + kbase + half * 8);
  v8bf a1 = *(const v8bf*)(row + kbase + 16 + half * 8);
  return cat(a0, a1);
}
__device__ __forceinline__ v16bf load_b_bf16(const __bf16* row, int kbase, int half) {
  v8bf b0 = *(const v8bf*)(row + kbase + half * 8);
  v8bf b1 = *(const v8bf*)(row + kbase + 16 + half * 8);
  return cat(b0, b1);
}

// ---- CDNA5 async global->LDS (ASYNCcnt path, ISA §10.7 / §15.18) -------------
__device__ __forceinline__ unsigned lds_off32(const void* p) {
  // generic LDS pointer: addr[31:0] is the LDS offset (aperture mapping, ISA 10.2)
  return (unsigned)(size_t)p;
}
__device__ __forceinline__ void async_ld16(unsigned lds_byte_off, const void* g) {
  asm volatile("global_load_async_to_lds_b128 %0, %1, off"
               :: "v"(lds_byte_off), "v"(g) : "memory");
}
__device__ __forceinline__ void wait_async0() {
  asm volatile("s_wait_asynccnt 0" ::: "memory");
}

// ---------------------------------------------------------------------------
// Kernel 0: one-shot weight transpose+convert: dst[c*R + r] = bf16(src[r*C + c])
// ---------------------------------------------------------------------------
__global__ __launch_bounds__(256) void transpose_bf16_kernel(
    const float* __restrict__ src, __bf16* __restrict__ dst, int R, int C) {
  int idx = blockIdx.x * 256 + threadIdx.x;
  if (idx < R * C) {
    int r = idx / C, c = idx % C;
    dst[(size_t)c * R + r] = (__bf16)src[idx];
  }
}

// ---------------------------------------------------------------------------
// Kernel 1: K/V projection. One workgroup per (b,h), 20 waves (5 s-tiles x 4 d-tiles).
// K stored [s][d] (B-matrix layout for QK^T), V stored transposed [d][s] (B for P@V).
// Pad rows are produced from clamped (in-bounds) loads and simply never stored.
// ---------------------------------------------------------------------------
__global__ __launch_bounds__(640) void kv_proj_kernel(
    const float* __restrict__ kv, const __bf16* __restrict__ WkT,
    const __bf16* __restrict__ WvT, __bf16* __restrict__ Kmat,
    __bf16* __restrict__ VT) {
  const int h = blockIdx.x, b = blockIdx.y;
  const int tid = threadIdx.x;
  const int wave = tid >> 5, lane = tid & 31;
  const int lane16 = lane & 15, half = lane >> 4;

  __bf16* kk = Kmat + (size_t)(b * HEADS + h) * SPAD * DIM_HEAD;  // [s][64]
  __bf16* vt = VT   + (size_t)(b * HEADS + h) * DIM_HEAD * SPAD;  // [64][s]

  // zero the s-pad regions (disjoint from tile stores)
  for (int i = tid; i < (SPAD - SKV) * DIM_HEAD; i += 640)
    kk[SKV * DIM_HEAD + i] = (__bf16)0.0f;
  for (int i = tid; i < DIM_HEAD * (SPAD - SKV); i += 640) {
    int d = i / (SPAD - SKV), s = SKV + i % (SPAD - SKV);
    vt[d * SPAD + s] = (__bf16)0.0f;
  }

  const int mt = wave >> 2;   // s tile 0..4
  const int nt = wave & 3;    // d tile 0..3
  const int s_lane = mt * 16 + lane16;
  const int sc = (s_lane < SKV) ? s_lane : (SKV - 1);   // clamp: pad rows are junk, never stored
  const float* Arow = kv + (size_t)b * SKV * DKV + (size_t)sc * DKV;
  const __bf16* BKrow = WkT + (size_t)(h * DIM_HEAD + nt * 16 + lane16) * DKV;
  const __bf16* BVrow = WvT + (size_t)(h * DIM_HEAD + nt * 16 + lane16) * DKV;

  v8f accK = {}, accV = {};
  for (int kc = 0; kc < DKV / 32; ++kc) {
    const float* p = Arow + kc * 32;
    float4 x0 = *(const float4*)(p + half * 8);
    float4 x1 = *(const float4*)(p + half * 8 + 4);
    float4 x2 = *(const float4*)(p + 16 + half * 8);
    float4 x3 = *(const float4*)(p + 16 + half * 8 + 4);
    v16bf a = cat(cvt8(x0, x1), cvt8(x2, x3));
    v16bf bk = load_b_bf16(BKrow, kc * 32, half);
    v16bf bv = load_b_bf16(BVrow, kc * 32, half);
    accK = wmma_bf16(a, bk, accK);
    accV = wmma_bf16(a, bv, accV);
  }
#pragma unroll
  for (int r = 0; r < 8; ++r) {
    int s = mt * 16 + r + half * 8;
    int d = nt * 16 + lane16;
    if (s < SKV) {
      kk[s * DIM_HEAD + d] = (__bf16)accK[r];
      vt[d * SPAD + s]     = (__bf16)accV[r];
    }
  }
}

// ---------------------------------------------------------------------------
// Kernel 2: Q projection. Block = 8 waves sharing one 16-row A tile staged in LDS
// (coalesced float4 reads of the [C][N] source), wave w owns 64-wide n-group w.
// ---------------------------------------------------------------------------
__global__ __launch_bounds__(256) void q_proj_kernel(
    const float* __restrict__ qin, const __bf16* __restrict__ WqT,
    __bf16* __restrict__ Q) {
  __shared__ __bf16 sA[16 * AST];
  const int tid = threadIdx.x;
  const int b = blockIdx.x >> 8;
  const int mt = blockIdx.x & 255;
  const int m0 = mt * 16;
  const float* qb = qin + (size_t)b * CCH * NQ;

  // stage A tile: 320 (c) x 16 (m), coalesced float4 along m
  for (int idx = tid; idx < CCH * 4; idx += 256) {
    int c = idx >> 2, mg = idx & 3;
    float4 v = *(const float4*)(qb + (size_t)c * NQ + m0 + mg * 4);
    sA[(mg * 4 + 0) * AST + c] = (__bf16)v.x;
    sA[(mg * 4 + 1) * AST + c] = (__bf16)v.y;
    sA[(mg * 4 + 2) * AST + c] = (__bf16)v.z;
    sA[(mg * 4 + 3) * AST + c] = (__bf16)v.w;
  }
  __syncthreads();

  const int wave = tid >> 5, lane = tid & 31;
  const int lane16 = lane & 15, half = lane >> 4;
  const int ng = wave;   // 0..7
  const __bf16* arow = sA + lane16 * AST;

  v8f acc[4] = {};
  for (int kc = 0; kc < CCH / 32; ++kc) {
    v16bf a = load_a_bf16(arow, kc * 32, half);
#pragma unroll
    for (int nt = 0; nt < 4; ++nt) {
      const __bf16* brow = WqT + (size_t)(ng * 64 + nt * 16 + lane16) * CCH;
      v16bf bb = load_b_bf16(brow, kc * 32, half);
      acc[nt] = wmma_bf16(a, bb, acc[nt]);
    }
  }
#pragma unroll
  for (int nt = 0; nt < 4; ++nt)
#pragma unroll
    for (int r = 0; r < 8; ++r) {
      int m = m0 + r + half * 8;
      int n = ng * 64 + nt * 16 + lane16;
      Q[((size_t)b * NQ + m) * INNER + n] = (__bf16)acc[nt][r];
    }
}

// ---------------------------------------------------------------------------
// Kernel 3: attention core. One workgroup per (b,h,128 queries), 8 waves.
// K [s][d] and V^T [d][s] staged to LDS via async b128 copies (ASYNCcnt).
// ---------------------------------------------------------------------------
__global__ __launch_bounds__(256) void attn_kernel(
    const __bf16* __restrict__ Q, const __bf16* __restrict__ Kmat,
    const __bf16* __restrict__ VT, __bf16* __restrict__ O) {
  __shared__ __bf16 sK[SPAD * SKST];       // [s][d], stride 72
  __shared__ __bf16 sV[DIM_HEAD * SVST];   // [d][s], stride 104
  __shared__ __bf16 sP[8][16 * PST];       // per-wave C->A transpose staging

  const int b = blockIdx.z, h = blockIdx.y, mb = blockIdx.x;
  const int tid = threadIdx.x;
  const __bf16* gK = Kmat + (size_t)(b * HEADS + h) * SPAD * DIM_HEAD;
  const __bf16* gV = VT   + (size_t)(b * HEADS + h) * DIM_HEAD * SPAD;

  // async global->LDS staging: 96x64 K (8 x 16B per row), 64x96 V^T (12 x 16B per row)
  for (int j = tid; j < SPAD * 8; j += 256) {          // 768 chunks
    int row = j >> 3, off = (j & 7) * 8;
    async_ld16(lds_off32(&sK[row * SKST + off]), gK + row * DIM_HEAD + off);
  }
  for (int j = tid; j < DIM_HEAD * 12; j += 256) {     // 768 chunks
    int row = j / 12, off = (j % 12) * 8;
    async_ld16(lds_off32(&sV[row * SVST + off]), gV + row * SPAD + off);
  }
  wait_async0();
  __syncthreads();

  const int wave = tid >> 5, lane = tid & 31;
  const int lane16 = lane & 15, half = lane >> 4;
  const int m0 = mb * 128 + wave * 16;
  const __bf16* gQ = Q + ((size_t)b * NQ + m0) * INNER + h * DIM_HEAD;

  // Q A-fragments, K-dim 64 (2 chunks), vectorized 16B loads
  const __bf16* qrow = gQ + (size_t)lane16 * INNER;
  v16bf aq0 = load_a_bf16(qrow, 0, half);
  v16bf aq1 = load_a_bf16(qrow, 32, half);

  // logits: 6 s-tiles x (K=64); B reads along d from sK rows (contiguous)
  v8f lg[6] = {};
#pragma unroll
  for (int st = 0; st < 6; ++st) {
    const __bf16* krow = sK + (st * 16 + lane16) * SKST;
    lg[st] = wmma_bf16(aq0, load_b_bf16(krow, 0, half), lg[st]);
    lg[st] = wmma_bf16(aq1, load_b_bf16(krow, 32, half), lg[st]);
  }

  // masked softmax over s (row m lives across one 16-lane half; xor<16 stays in-half)
  const float scale = 0.125f;   // 64^-0.5
  float p[6][8];
#pragma unroll
  for (int r = 0; r < 8; ++r) {
    float x[6], rmax = -1e30f;
#pragma unroll
    for (int st = 0; st < 6; ++st) {
      int s = st * 16 + lane16;
      x[st] = (s < SKV) ? lg[st][r] * scale : -1e30f;
      rmax = fmaxf(rmax, x[st]);
    }
#pragma unroll
    for (int off = 1; off <= 8; off <<= 1)
      rmax = fmaxf(rmax, __shfl_xor(rmax, off, 32));
    float rsum = 0.0f;
#pragma unroll
    for (int st = 0; st < 6; ++st) {
      float e = __expf(x[st] - rmax);   // masked entries underflow to 0
      p[st][r] = e;
      rsum += e;
    }
#pragma unroll
    for (int off = 1; off <= 8; off <<= 1)
      rsum += __shfl_xor(rsum, off, 32);
    float rinv = 1.0f / rsum;
#pragma unroll
    for (int st = 0; st < 6; ++st) p[st][r] *= rinv;
  }

  // transpose P (C-layout) -> A-layout through LDS
#pragma unroll
  for (int st = 0; st < 6; ++st)
#pragma unroll
    for (int r = 0; r < 8; ++r)
      sP[wave][(r + half * 8) * PST + st * 16 + lane16] = (__bf16)p[st][r];
  __syncthreads();

  // out = P @ V : K = 96 (3 chunks), N = 64 (4 d-tiles); B reads along s from sV rows
  const __bf16* prow = sP[wave] + lane16 * PST;
  v8f o[4] = {};
#pragma unroll
  for (int kc = 0; kc < 3; ++kc) {
    v16bf ap = load_a_bf16(prow, kc * 32, half);
#pragma unroll
    for (int dt = 0; dt < 4; ++dt) {
      const __bf16* vrow = sV + (dt * 16 + lane16) * SVST;
      o[dt] = wmma_bf16(ap, load_b_bf16(vrow, kc * 32, half), o[dt]);
    }
  }

  __bf16* gO = O + ((size_t)b * NQ + m0) * INNER + h * DIM_HEAD;
#pragma unroll
  for (int dt = 0; dt < 4; ++dt)
#pragma unroll
    for (int r = 0; r < 8; ++r)
      gO[(size_t)(r + half * 8) * INNER + dt * 16 + lane16] = (__bf16)o[dt][r];
}

// ---------------------------------------------------------------------------
// Kernel 4: output projection [4096x512]x[512x320] + bo, fp32 out in [B,C,N].
// ---------------------------------------------------------------------------
__global__ __launch_bounds__(128) void out_proj_kernel(
    const __bf16* __restrict__ O, const __bf16* __restrict__ WoT,
    const float* __restrict__ bo, float* __restrict__ y) {
  const int wid = blockIdx.x * 4 + (threadIdx.x >> 5);
  const int lane = threadIdx.x & 31, lane16 = lane & 15, half = lane >> 4;
  const int b = wid / 1280;          // 256 m-tiles * 5 n-groups
  const int rem = wid % 1280;
  const int mt = rem / 5;
  const int ng = rem % 5;            // 64-wide n group (320 = 5*64)
  const int m0 = mt * 16;
  const __bf16* arow = O + ((size_t)b * NQ + m0 + lane16) * INNER;

  v8f acc[4] = {};
  for (int kc = 0; kc < INNER / 32; ++kc) {
    v16bf a = load_a_bf16(arow, kc * 32, half);
#pragma unroll
    for (int nt = 0; nt < 4; ++nt) {
      const __bf16* brow = WoT + (size_t)(ng * 64 + nt * 16 + lane16) * INNER;
      v16bf bb = load_b_bf16(brow, kc * 32, half);
      acc[nt] = wmma_bf16(a, bb, acc[nt]);
    }
  }
#pragma unroll
  for (int nt = 0; nt < 4; ++nt)
#pragma unroll
    for (int r = 0; r < 8; ++r) {
      int c = ng * 64 + nt * 16 + lane16;
      int m = m0 + r + half * 8;
      y[((size_t)b * CCH + c) * NQ + m] = acc[nt][r] + bo[c];
    }
}

// ---------------------------------------------------------------------------
extern "C" void kernel_launch(void* const* d_in, const int* in_sizes, int n_in,
                              void* d_out, int out_size, void* d_ws, size_t ws_size,
                              hipStream_t stream) {
  const float* qin = (const float*)d_in[0];   // [B,C,H,W]
  const float* kvp = (const float*)d_in[1];   // [B,77,1024]
  const float* Wq  = (const float*)d_in[2];   // [320,512]
  const float* Wk  = (const float*)d_in[3];   // [1024,512]
  const float* Wv  = (const float*)d_in[4];   // [1024,512]
  const float* Wo  = (const float*)d_in[5];   // [512,320]
  const float* bo  = (const float*)d_in[6];   // [320]
  float* y = (float*)d_out;

  char* ws = (char*)d_ws;
  const size_t qBytes = (size_t)BATCH * NQ * INNER * sizeof(__bf16);        // 32 MB
  const size_t kvBytes = (size_t)BATCH * HEADS * SPAD * DIM_HEAD * sizeof(__bf16);
  __bf16* Q    = (__bf16*)ws;                       ws += qBytes;
  __bf16* O    = (__bf16*)ws;                       ws += qBytes;
  __bf16* Kmat = (__bf16*)ws;                       ws += kvBytes;
  __bf16* VT   = (__bf16*)ws;                       ws += kvBytes;
  __bf16* WqT  = (__bf16*)ws;                       ws += (size_t)CCH * INNER * 2;
  __bf16* WkT  = (__bf16*)ws;                       ws += (size_t)DKV * INNER * 2;
  __bf16* WvT  = (__bf16*)ws;                       ws += (size_t)DKV * INNER * 2;
  __bf16* WoT  = (__bf16*)ws;                       ws += (size_t)INNER * CCH * 2;

  transpose_bf16_kernel<<<(CCH * INNER + 255) / 256, 256, 0, stream>>>(Wq, WqT, CCH, INNER);
  transpose_bf16_kernel<<<(DKV * INNER + 255) / 256, 256, 0, stream>>>(Wk, WkT, DKV, INNER);
  transpose_bf16_kernel<<<(DKV * INNER + 255) / 256, 256, 0, stream>>>(Wv, WvT, DKV, INNER);
  transpose_bf16_kernel<<<(INNER * CCH + 255) / 256, 256, 0, stream>>>(Wo, WoT, INNER, CCH);

  kv_proj_kernel<<<dim3(HEADS, BATCH), 640, 0, stream>>>(kvp, WkT, WvT, Kmat, VT);
  q_proj_kernel<<<BATCH * 256, 256, 0, stream>>>(qin, WqT, Q);
  attn_kernel<<<dim3(NQ / 128, HEADS, BATCH), 256, 0, stream>>>(Q, Kmat, VT, O);
  out_proj_kernel<<<(BATCH * 256 * 5) / 4, 128, 0, stream>>>(O, WoT, bo, y);
}
